// Model_17008070493068
// MI455X (gfx1250) — matrix operand order repaired and checked by reading, not compiled, not run on hardware
//
#include <hip/hip_runtime.h>
#include <hip/hip_bf16.h>

// ---------------------------------------------------------------------------
// GRU seq2seq w/ attention on gfx1250.  B=16, S=48, HID=150(->160), GH=256,
// vocab=32000.  GEMMs: v_wmma_f32_16x16x32_f16 (M tile = batch = 16).
// GRU gate-input tiles staged LDS via TDM tensor_load_to_lds when available.
// ---------------------------------------------------------------------------

typedef _Float16 v16h __attribute__((ext_vector_type(16)));
typedef _Float16 h8   __attribute__((ext_vector_type(8)));
typedef float    v8f  __attribute__((ext_vector_type(8)));

#define WMMA_F16(a,b,c) \
  __builtin_amdgcn_wmma_f32_16x16x32_f16(false,(a),false,(b),(short)0,(c),false,false)

#if __has_builtin(__builtin_amdgcn_tensor_load_to_lds) && \
    __has_builtin(__builtin_amdgcn_s_wait_tensorcnt)
#define HAVE_TDM 1
#else
#define HAVE_TDM 0
#endif

#if HAVE_TDM
typedef unsigned int u32x4 __attribute__((ext_vector_type(4)));
typedef int          i32x8 __attribute__((ext_vector_type(8)));
typedef int          i32x4 __attribute__((ext_vector_type(4)));

// TDM DMA: copy a 16x768 f32 tile (row stride 768) from global into LDS.
// D# group0: count=1, lds_addr, global_addr[56:0], type=2.
// D# group1: data_size=4B, tensor_dim0=768, tensor_dim1=16, tile 768x16,
//            tensor_dim0_stride=768.
__device__ __forceinline__ void tdm_load_gx_tile(const float* gsrc,
                                                 unsigned lds_byte_addr) {
  const unsigned long long ga = (unsigned long long)(size_t)gsrc;
  u32x4 g0;
  g0[0] = 1u;                                               // count=1
  g0[1] = lds_byte_addr;                                    // lds_addr
  g0[2] = (unsigned)(ga & 0xFFFFFFFFu);                     // global_addr[31:0]
  g0[3] = (unsigned)((ga >> 32) & 0x1FFFFFFu) | (2u << 30); // addr[56:32]|type=2
  i32x8 g1;
  g1[0] = (2 << 16);            // workgroup_mask=0, data_size=4B
  g1[1] = (768 & 0xFFFF) << 16; // tensor_dim0[15:0]
  g1[2] = (16 << 16);           // tensor_dim0[31:16]=0, tensor_dim1[15:0]=16
  g1[3] = (768 << 16);          // tensor_dim1[31:16]=0, tile_dim0=768
  g1[4] = 16;                   // tile_dim1=16, tile_dim2=0
  g1[5] = 768;                  // tensor_dim0_stride[31:0]
  g1[6] = 0;                    // stride0[47:32]=0, dim1_stride[15:0]=0
  g1[7] = 0;
  i32x4 z4 = {0, 0, 0, 0};
#if __clang_major__ >= 23
  i32x8 z8 = {0, 0, 0, 0, 0, 0, 0, 0};
  __builtin_amdgcn_tensor_load_to_lds(g0, g1, z4, z4, z8, 0);
#else
  __builtin_amdgcn_tensor_load_to_lds(g0, g1, z4, z4, 0);
#endif
}
#endif

// A fragment (16x32 f16, row-major, stride lda halves).
__device__ __forceinline__ v16h load_a_frag(const _Float16* A, int lda,
                                            int mbase, int kk, int lane) {
  const int m   = mbase + (lane & 15);
  const int khi = (lane >> 4) & 1;
  const _Float16* p = A + (size_t)m * lda + kk + khi * 8;
  h8 lo = *(const h8*)(p);
  h8 hi = *(const h8*)(p + 16);
  v16h a;
#pragma unroll
  for (int i = 0; i < 8; ++i) { a[i] = lo[i]; a[8 + i] = hi[i]; }
  return a;
}

// B fragment (32x16 f16) from W stored row-major (N x K): B[k][n] = W[n][k].
__device__ __forceinline__ v16h load_b_frag(const _Float16* W, int ldw,
                                            int nbase, int kk, int lane) {
  const int n   = nbase + (lane & 15);
  const int khi = (lane >> 4) & 1;
  const _Float16* p = W + (size_t)n * ldw + kk + khi * 8;
  h8 lo = *(const h8*)(p);
  h8 hi = *(const h8*)(p + 16);
  v16h b;
#pragma unroll
  for (int i = 0; i < 8; ++i) { b[i] = lo[i]; b[8 + i] = hi[i]; }
  return b;
}

// ---------------------------------------------------------------------------
// C[M,N] = A[M,K] @ W[N,K]^T (+bias) -> f32 and/or f16.  Software-pipelined
// K loop; each wave owns NT consecutive 16x16 N-tiles sharing one A fragment.
// ---------------------------------------------------------------------------
template <int NT>
__global__ __launch_bounds__(256) void gemm_xwT_kernel(
    const _Float16* __restrict__ A, int lda,
    const _Float16* __restrict__ W, int ldw,
    const float* __restrict__ bias, int nbias,
    float* __restrict__ Cf, long long ldc, long long cofs,
    _Float16* __restrict__ Ch, int ldch,
    int N, int K) {
  const int lane = threadIdx.x & 31;
  const int wave = threadIdx.x >> 5;
  const int ntiles = N >> 4;
  const int nt0 = (blockIdx.x * 8 + wave) * NT;
  if (nt0 >= ntiles) return;
  const int mbase = blockIdx.y << 4;

  const v8f vzero = {};
  v8f acc[NT];
#pragma unroll
  for (int j = 0; j < NT; ++j) acc[j] = vzero;

  v16h a_cur = load_a_frag(A, lda, mbase, 0, lane);
  v16h b_cur[NT];
#pragma unroll
  for (int j = 0; j < NT; ++j) b_cur[j] = load_b_frag(W, ldw, (nt0 + j) << 4, 0, lane);

  for (int kk = 0; kk < K; kk += 32) {
    v16h a_nxt = a_cur;
    v16h b_nxt[NT];
#pragma unroll
    for (int j = 0; j < NT; ++j) b_nxt[j] = b_cur[j];
    if (kk + 32 < K) {               // preload next K block -> overlap w/ WMMA
      a_nxt = load_a_frag(A, lda, mbase, kk + 32, lane);
#pragma unroll
      for (int j = 0; j < NT; ++j)
        b_nxt[j] = load_b_frag(W, ldw, (nt0 + j) << 4, kk + 32, lane);
      __builtin_prefetch((const void*)(W + (size_t)(nt0 << 4) * ldw + kk + 64), 0, 1);
    }
#pragma unroll
    for (int j = 0; j < NT; ++j) acc[j] = WMMA_F16(a_cur, b_cur[j], acc[j]);
    a_cur = a_nxt;
#pragma unroll
    for (int j = 0; j < NT; ++j) b_cur[j] = b_nxt[j];
  }

  const int mofs = mbase + ((lane >> 4) << 3);
#pragma unroll
  for (int j = 0; j < NT; ++j) {
    const int nb = (nt0 + j) << 4;
    if (nb < N) {
      const int n = nb + (lane & 15);
      const float bv = (bias && n < nbias) ? bias[n] : 0.f;
#pragma unroll
      for (int i = 0; i < 8; ++i) {
        const float v = acc[j][i] + bv;
        const int m = mofs + i;
        if (Cf) Cf[cofs + (long long)m * ldc + n] = v;
        if (Ch) Ch[(long long)m * ldch + n] = (_Float16)v;
      }
    }
  }
}

// ---------------------------------------------------------------------------
// Persistent GRU time-scan (one workgroup, 8 waves).  gh = h@Whh^T via WMMA;
// gate-input tile gx[t] (16x768 f32) DMA'd into LDS by the TDM while the
// WMMA phase runs; gate math in f32; h carried in regs (+f16 LDS mirror).
// ---------------------------------------------------------------------------
__global__ __launch_bounds__(256) void gru_scan_kernel(
    const _Float16* __restrict__ Whh,   // 768 x 256 f16
    const float* __restrict__ bhh,      // 768
    const float* __restrict__ gx,       // (T*16) x 768 f32, row = t*16+b
    float* __restrict__ hstate,         // 16 x 256 f32 (in/out)
    _Float16* __restrict__ y, int ldy, int ycol,
    _Float16* __restrict__ hfin, int ldh, int hcol,
    int T, int rev) {
  __shared__ float gh[16 * 768];
  __shared__ __align__(16) _Float16 hh[16 * 256];
#if HAVE_TDM
  __shared__ __align__(16) float gx_lds[16 * 768];
#endif
  const int tid  = threadIdx.x;
  const int lane = tid & 31;
  const int wave = tid >> 5;
  float hreg[16];
#pragma unroll
  for (int b = 0; b < 16; ++b) {
    const float v = hstate[b * 256 + tid];
    hreg[b] = v;
    hh[b * 256 + tid] = (_Float16)v;
  }
  __syncthreads();
  for (int step = 0; step < T; ++step) {
    const int t = rev ? (T - 1 - step) : step;
#if HAVE_TDM
    if (wave == 0)                       // DMA gate inputs while WMMA runs
      tdm_load_gx_tile(gx + (size_t)t * 16 * 768,
                       (unsigned)(size_t)(void*)gx_lds);
#endif
    for (int nt = wave; nt < 48; nt += 8) {
      v8f acc = {};
      v16h b_cur = load_b_frag(Whh, 256, nt << 4, 0, lane);
      for (int kk = 0; kk < 256; kk += 32) {
        v16h a = load_a_frag((const _Float16*)hh, 256, 0, kk, lane);
        v16h b_nxt = b_cur;
        if (kk + 32 < 256) b_nxt = load_b_frag(Whh, 256, nt << 4, kk + 32, lane);
        acc = WMMA_F16(a, b_cur, acc);
        b_cur = b_nxt;
      }
      const int n = (nt << 4) + (lane & 15);
      const int mofs = (lane >> 4) << 3;
#pragma unroll
      for (int i = 0; i < 8; ++i) gh[(mofs + i) * 768 + n] = acc[i];
    }
#if HAVE_TDM
    if (wave == 0) __builtin_amdgcn_s_wait_tensorcnt(0);
#endif
    __syncthreads();
#if HAVE_TDM
    const float* gxr = gx_lds;
#else
    const float* gxr = gx + (size_t)t * 16 * 768;
#endif
    const float br = bhh[tid], bz = bhh[256 + tid], bn = bhh[512 + tid];
    for (int b = 0; b < 16; ++b) {
      const float* gxb = gxr + b * 768;
      const float* ghb = gh  + b * 768;
      const float r  = 1.f / (1.f + __expf(-(gxb[tid]       + ghb[tid]       + br)));
      const float z  = 1.f / (1.f + __expf(-(gxb[256 + tid] + ghb[256 + tid] + bz)));
      const float nn = tanhf(gxb[512 + tid] + r * (ghb[512 + tid] + bn));
      const float hn = (1.f - z) * nn + z * hreg[b];
      hreg[b] = hn;
      hh[b * 256 + tid] = (_Float16)hn;
      if (y) y[(size_t)(t * 16 + b) * ldy + ycol + tid] = (_Float16)hn;
    }
    __syncthreads();
  }
#pragma unroll
  for (int b = 0; b < 16; ++b) {
    hstate[b * 256 + tid] = hreg[b];
    if (hfin) hfin[(size_t)b * ldh + hcol + tid] = (_Float16)hreg[b];
  }
}

// ---------------------------------------------------------------------------
// Attention for one outer step (scores, softmax over axis=l, weighted sum).
// ---------------------------------------------------------------------------
__global__ __launch_bounds__(256) void attn_kernel(
    const float* __restrict__ Qm, const float* __restrict__ Km,
    const float* __restrict__ Vm, _Float16* __restrict__ data_h, int L) {
  __shared__ float sc[48 * 48];
  const int b = blockIdx.x, tid = threadIdx.x;
  for (int i = tid; i < L * L; i += 256) {
    const int l = i / L, m = i - l * L;
    const float* q = Qm + (size_t)(l * 16 + b) * 160;
    const float* k = Km + (size_t)(m * 16 + b) * 160;
    float s = 0.f;
    for (int d = 0; d < 150; ++d) s += q[d] * k[d];
    sc[l * 48 + m] = s;
  }
  __syncthreads();
  for (int m = tid; m < L; m += 256) {
    float mx = -3.4e38f;
    for (int l = 0; l < L; ++l) mx = fmaxf(mx, sc[l * 48 + m]);
    float sum = 0.f;
    for (int l = 0; l < L; ++l) { float e = __expf(sc[l * 48 + m] - mx); sc[l * 48 + m] = e; sum += e; }
    const float inv = 1.f / sum;
    for (int l = 0; l < L; ++l) sc[l * 48 + m] *= inv;
  }
  __syncthreads();
  for (int i = tid; i < L * 160; i += 256) {
    const int l = i / 160, d = i - l * 160;
    float s = 0.f;
    if (d < 150)
      for (int m = 0; m < L; ++m) s += sc[l * 48 + m] * Vm[(size_t)(m * 16 + b) * 160 + d];
    data_h[(size_t)(l * 16 + b) * 160 + d] = (_Float16)s;
  }
}

// ---------------------------------------------------------------------------
// Helpers
// ---------------------------------------------------------------------------
__global__ void cast_pad_kernel(const float* __restrict__ src, _Float16* __restrict__ dst,
                                int srows, int scols, int drows, int dcols) {
  const int i = blockIdx.x * 256 + threadIdx.x;
  if (i >= drows * dcols) return;
  const int r = i / dcols, c = i - r * dcols;
  dst[i] = (r < srows && c < scols) ? (_Float16)src[(size_t)r * scols + c] : (_Float16)0.f;
}

__global__ void embed_x_kernel(const int* __restrict__ x, const float* __restrict__ emb,
                               _Float16* __restrict__ dst) {
  const int i = blockIdx.x * 256 + threadIdx.x;
  if (i >= 48 * 16 * 160) return;
  const int d = i % 160, r = i / 160, b = r & 15, s = r >> 4;
  _Float16 v = (_Float16)0.f;
  if (d < 150) v = (_Float16)emb[(size_t)x[b * 48 + s] * 150 + d];
  dst[i] = v;
}

__global__ void embed_it_kernel(const int* __restrict__ target, const float* __restrict__ emb,
                                _Float16* __restrict__ dst) {
  const int i = blockIdx.x * 256 + threadIdx.x;
  if (i >= 48 * 16 * 160) return;
  const int d = i % 160, r = i / 160, b = r & 15, s = r >> 4;
  _Float16 v = (_Float16)0.f;
  if (s > 0 && d < 150) v = (_Float16)emb[(size_t)target[b * 48 + (s - 1)] * 150 + d];
  dst[i] = v;
}

__global__ void fill_kernel(float* __restrict__ p, int n, float v) {
  const int i = blockIdx.x * 256 + threadIdx.x;
  if (i < n) p[i] = v;
}

__global__ void conv_combine_kernel(const float* __restrict__ hf0, const float* __restrict__ hb0,
                                    const float* __restrict__ hf1, const float* __restrict__ hb1,
                                    const float* __restrict__ cw, const float* __restrict__ cb,
                                    float* __restrict__ hd0, float* __restrict__ hd1) {
  const int i = blockIdx.x * 256 + threadIdx.x;
  if (i >= 16 * 256) return;
  const float e0 = hf0[i], e1 = hb0[i], e2 = hf1[i], e3 = hb1[i];
  hd0[i] = cw[0] * e0 + cw[1] * e1 + cw[2] * e2 + cw[3] * e3 + cb[0];
  hd1[i] = cw[4] * e0 + cw[5] * e1 + cw[6] * e2 + cw[7] * e3 + cb[1];
}

// ---------------------------------------------------------------------------
// Host orchestration
// ---------------------------------------------------------------------------
extern "C" void kernel_launch(void* const* d_in, const int* in_sizes, int n_in,
                              void* d_out, int out_size, void* d_ws, size_t ws_size,
                              hipStream_t stream) {
  (void)in_sizes; (void)n_in; (void)out_size; (void)ws_size;
  const int S = 48;

  const int*   x      = (const int*)d_in[0];
  const int*   target = (const int*)d_in[1];
  const float* emb    = (const float*)d_in[2];
  const float *W0f_ih = (const float*)d_in[3],  *W0f_hh = (const float*)d_in[4],
              *b0f_ih = (const float*)d_in[5],  *b0f_hh = (const float*)d_in[6];
  const float *W0b_ih = (const float*)d_in[7],  *W0b_hh = (const float*)d_in[8],
              *b0b_ih = (const float*)d_in[9],  *b0b_hh = (const float*)d_in[10];
  const float *W1f_ih = (const float*)d_in[11], *W1f_hh = (const float*)d_in[12],
              *b1f_ih = (const float*)d_in[13], *b1f_hh = (const float*)d_in[14];
  const float *W1b_ih = (const float*)d_in[15], *W1b_hh = (const float*)d_in[16],
              *b1b_ih = (const float*)d_in[17], *b1b_hh = (const float*)d_in[18];
  const float *Wd0_ih = (const float*)d_in[19], *Wd0_hh = (const float*)d_in[20],
              *bd0_ih = (const float*)d_in[21], *bd0_hh = (const float*)d_in[22];
  const float *Wd1_ih = (const float*)d_in[23], *Wd1_hh = (const float*)d_in[24],
              *bd1_ih = (const float*)d_in[25], *bd1_hh = (const float*)d_in[26];
  const float *convw  = (const float*)d_in[27], *convb  = (const float*)d_in[28];
  const float *Wo  = (const float*)d_in[29], *bo   = (const float*)d_in[30];
  const float *Wout= (const float*)d_in[31], *bout = (const float*)d_in[32];
  const float *WQ  = (const float*)d_in[33], *bQ   = (const float*)d_in[34];
  const float *WK  = (const float*)d_in[35], *bK   = (const float*)d_in[36];
  const float *WV  = (const float*)d_in[37], *bV   = (const float*)d_in[38];
  float* out = (float*)d_out;

  uint8_t* base = (uint8_t*)d_ws;
  size_t cur = 0;
  auto alloc = [&](size_t bytes) -> void* {
    void* p = base + cur;
    cur += (bytes + 255) & ~(size_t)255;
    return p;
  };

  _Float16* w0f_ih = (_Float16*)alloc(768 * 160 * 2);
  _Float16* w0f_hh = (_Float16*)alloc(768 * 256 * 2);
  _Float16* w0b_ih = (_Float16*)alloc(768 * 160 * 2);
  _Float16* w0b_hh = (_Float16*)alloc(768 * 256 * 2);
  _Float16* w1f_ih = (_Float16*)alloc(768 * 512 * 2);
  _Float16* w1f_hh = (_Float16*)alloc(768 * 256 * 2);
  _Float16* w1b_ih = (_Float16*)alloc(768 * 512 * 2);
  _Float16* w1b_hh = (_Float16*)alloc(768 * 256 * 2);
  _Float16* wd0_ih = (_Float16*)alloc(768 * 160 * 2);
  _Float16* wd0_hh = (_Float16*)alloc(768 * 256 * 2);
  _Float16* wd1_ih = (_Float16*)alloc(768 * 256 * 2);
  _Float16* wd1_hh = (_Float16*)alloc(768 * 256 * 2);
  _Float16* w_o    = (_Float16*)alloc(512 * 160 * 2);
  _Float16* w_q    = (_Float16*)alloc(160 * 512 * 2);
  _Float16* w_k    = (_Float16*)alloc(160 * 512 * 2);
  _Float16* w_v    = (_Float16*)alloc(160 * 512 * 2);
  _Float16* w_out  = (_Float16*)alloc((size_t)32000 * 512 * 2);
  _Float16* dt_h   = (_Float16*)alloc(768 * 160 * 2);
  _Float16* it_h   = (_Float16*)alloc(768 * 160 * 2);
  float*    gx0f   = (float*)alloc(768 * 768 * 4);
  float*    gx0b   = (float*)alloc(768 * 768 * 4);
  _Float16* x1_h   = (_Float16*)alloc(768 * 512 * 2);
  float*    gx1f   = (float*)alloc(768 * 768 * 4);
  float*    gx1b   = (float*)alloc(768 * 768 * 4);
  float*    hf0 = (float*)alloc(16 * 256 * 4);
  float*    hb0 = (float*)alloc(16 * 256 * 4);
  float*    hf1 = (float*)alloc(16 * 256 * 4);
  float*    hb1 = (float*)alloc(16 * 256 * 4);
  float*    hd0 = (float*)alloc(16 * 256 * 4);
  float*    hd1 = (float*)alloc(16 * 256 * 4);
  _Float16* ko_h   = (_Float16*)alloc(768 * 512 * 2);
  float*    Qf     = (float*)alloc(768 * 160 * 4);
  float*    Kf     = (float*)alloc(768 * 160 * 4);
  float*    Vf     = (float*)alloc(768 * 160 * 4);
  _Float16* data_h = (_Float16*)alloc(768 * 160 * 2);
  float*    gxd0   = (float*)alloc(768 * 768 * 4);
  _Float16* y0_h   = (_Float16*)alloc(768 * 256 * 2);
  float*    gxd1   = (float*)alloc(768 * 768 * 4);
  _Float16* endt2  = (_Float16*)alloc(16 * 512 * 2);

  auto cast = [&](const float* s, _Float16* d, int sr, int sc, int dr, int dc) {
    const int total = dr * dc;
    cast_pad_kernel<<<(total + 255) / 256, 256, 0, stream>>>(s, d, sr, sc, dr, dc);
  };
  auto gemm = [&](const _Float16* A, int lda, const _Float16* Wm, int ldw,
                  const float* bias, int nbias, float* Cf, long long ldc, long long cofs,
                  _Float16* Ch, int ldch, int M, int N, int K) {
    const int wtiles = (N / 16 + 1) / 2;                 // NT=2 tiles per wave
    dim3 grid((unsigned)((wtiles + 7) / 8), (unsigned)(M / 16));
    gemm_xwT_kernel<2><<<grid, 256, 0, stream>>>(A, lda, Wm, ldw, bias, nbias,
                                                 Cf, ldc, cofs, Ch, ldch, N, K);
  };
  auto zero = [&](float* p, int n) {
    fill_kernel<<<(n + 255) / 256, 256, 0, stream>>>(p, n, 0.f);
  };

  // weight conversion (f32 -> f16, K padded to 32, N padded to 16)
  cast(W0f_ih, w0f_ih, 768, 150, 768, 160);  cast(W0f_hh, w0f_hh, 768, 256, 768, 256);
  cast(W0b_ih, w0b_ih, 768, 150, 768, 160);  cast(W0b_hh, w0b_hh, 768, 256, 768, 256);
  cast(W1f_ih, w1f_ih, 768, 512, 768, 512);  cast(W1f_hh, w1f_hh, 768, 256, 768, 256);
  cast(W1b_ih, w1b_ih, 768, 512, 768, 512);  cast(W1b_hh, w1b_hh, 768, 256, 768, 256);
  cast(Wd0_ih, wd0_ih, 768, 150, 768, 160);  cast(Wd0_hh, wd0_hh, 768, 256, 768, 256);
  cast(Wd1_ih, wd1_ih, 768, 256, 768, 256);  cast(Wd1_hh, wd1_hh, 768, 256, 768, 256);
  cast(Wo,   w_o,   512, 150, 512, 160);
  cast(WQ,   w_q,   150, 512, 160, 512);
  cast(WK,   w_k,   150, 512, 160, 512);
  cast(WV,   w_v,   150, 512, 160, 512);
  cast(Wout, w_out, 32000, 512, 32000, 512);

  // embeddings (rows r = s*16+b)
  embed_x_kernel<<<(48 * 16 * 160 + 255) / 256, 256, 0, stream>>>(x, emb, dt_h);
  embed_it_kernel<<<(48 * 16 * 160 + 255) / 256, 256, 0, stream>>>(target, emb, it_h);

  // encoder layer 0 (bidirectional)
  gemm(dt_h, 160, w0f_ih, 160, b0f_ih, 768, gx0f, 768, 0, nullptr, 0, 768, 768, 160);
  gemm(dt_h, 160, w0b_ih, 160, b0b_ih, 768, gx0b, 768, 0, nullptr, 0, 768, 768, 160);
  zero(hf0, 4096); zero(hb0, 4096); zero(hf1, 4096); zero(hb1, 4096);
  gru_scan_kernel<<<1, 256, 0, stream>>>(w0f_hh, b0f_hh, gx0f, hf0,
                                         x1_h, 512, 0,   nullptr, 0, 0, S, 0);
  gru_scan_kernel<<<1, 256, 0, stream>>>(w0b_hh, b0b_hh, gx0b, hb0,
                                         x1_h, 512, 256, nullptr, 0, 0, S, 1);

  // encoder layer 1 (bidirectional; only final h needed)
  gemm(x1_h, 512, w1f_ih, 512, b1f_ih, 768, gx1f, 768, 0, nullptr, 0, 768, 768, 512);
  gemm(x1_h, 512, w1b_ih, 512, b1b_ih, 768, gx1b, 768, 0, nullptr, 0, 768, 768, 512);
  gru_scan_kernel<<<1, 256, 0, stream>>>(w1f_hh, b1f_hh, gx1f, hf1,
                                         nullptr, 0, 0, nullptr, 0, 0, S, 0);
  gru_scan_kernel<<<1, 256, 0, stream>>>(w1b_hh, b1b_hh, gx1b, hb1,
                                         nullptr, 0, 0, nullptr, 0, 0, S, 1);

  // decoder initial hiddens: 1x1 conv over stacked finals
  conv_combine_kernel<<<16, 256, 0, stream>>>(hf0, hb0, hf1, hb1, convw, convb, hd0, hd1);

  // attention projections for all 48 itinfo rows at once
  gemm(it_h, 160, w_o, 160, bo, 512, nullptr, 0, 0, ko_h, 512, 768, 512, 160);
  gemm(ko_h, 512, w_q, 512, bQ, 150, Qf, 160, 0, nullptr, 0, 768, 160, 512);
  gemm(ko_h, 512, w_k, 512, bK, 150, Kf, 160, 0, nullptr, 0, 768, 160, 512);
  gemm(ko_h, 512, w_v, 512, bV, 150, Vf, 160, 0, nullptr, 0, 768, 160, 512);

  // decoder loop
  const long long ldout = (long long)S * 32000;
  for (int idx = 0; idx < S; ++idx) {
    const int L = idx + 1;
    attn_kernel<<<16, 256, 0, stream>>>(Qf, Kf, Vf, data_h, L);
    gemm(data_h, 160, wd0_ih, 160, bd0_ih, 768, gxd0, 768, 0, nullptr, 0, L * 16, 768, 160);
    gru_scan_kernel<<<1, 256, 0, stream>>>(wd0_hh, bd0_hh, gxd0, hd0,
                                           y0_h, 256, 0, endt2, 512, 0,   L, 0);
    gemm(y0_h, 256, wd1_ih, 256, bd1_ih, 768, gxd1, 768, 0, nullptr, 0, L * 16, 768, 256);
    gru_scan_kernel<<<1, 256, 0, stream>>>(wd1_hh, bd1_hh, gxd1, hd1,
                                           nullptr, 0, 0, endt2, 512, 256, L, 0);
    gemm(endt2, 512, w_out, 512, bout, 32000,
         out, ldout, (long long)idx * 32000, nullptr, 0, 16, 32000, 512);
  }
}
